// IntrospectiveAlignmentLayer_23639499997471
// MI455X (gfx1250) — matrix operand order repaired and checked by reading, not compiled, not run on hardware
//
#include <hip/hip_runtime.h>

// ---------------------------------------------------------------------------
// IntrospectiveAlignmentLayer for MI455X (gfx1250, wave32, WMMA bf16)
//  - all GEMMs: bf16 A/B, f32 accum via v_wmma_f32_16x16x32_bf16
//    128x128x64 block tile, 8 waves * 16 WMMA per stage, double-buffered LDS
//    fed by GLOBAL_LOAD_ASYNC_TO_LDS_B128 when the toolchain exposes it
//  - band-masked self-attention computed analytically (mask * S => off-band
//    scores are exactly 0, handled in closed form; no dense 2048x2048 GEMM)
//  - LSTM input projections hoisted out of the sequential scan as big GEMMs
// ---------------------------------------------------------------------------

typedef __bf16 bf16;
typedef __attribute__((ext_vector_type(16))) __bf16 v16bf;
typedef __attribute__((ext_vector_type(8)))  __bf16 v8bf;
typedef __attribute__((ext_vector_type(8)))  float  v8f;
typedef __attribute__((ext_vector_type(4)))  float  v4f;
typedef __attribute__((ext_vector_type(4)))  int    v4i;

constexpr int cD  = 256;
constexpr int cLC = 2048;
constexpr int cLQ = 64;
constexpr int cBZ = 4;
constexpr int cNB = 32;   // band half width
constexpr int cNL = 5;
constexpr int cHD = 4 * cD;   // 1024

#if __has_builtin(__builtin_amdgcn_global_load_async_to_lds_b128) && \
    __has_builtin(__builtin_amdgcn_s_wait_asynccnt)
#define HAS_ASYNC_LDS 1
#else
#define HAS_ASYNC_LDS 0
#endif

// builtin wants v4i pointers in AS1 (global) / AS3 (LDS); build them via
// integer casts (generic LDS ptr low 32 bits == LDS byte offset per CDNA5
// aperture rules; global generic bits == AS1 bits)
#define AS1P(p) ((__attribute__((address_space(1))) v4i*)(unsigned long long)(const void*)(p))
#define AS3P(p) ((__attribute__((address_space(3))) v4i*)(unsigned)(unsigned long long)(const void*)(p))

__device__ __forceinline__ bf16 f2bf(float f) {
  union { float f; unsigned u; } v; v.f = f;
  unsigned r = v.u + 0x7FFFu + ((v.u >> 16) & 1u);   // round-nearest-even
  unsigned short h = (unsigned short)(r >> 16);
  bf16 b; __builtin_memcpy(&b, &h, 2); return b;
}
__device__ __forceinline__ float bf2f(bf16 b) {
  unsigned short h; __builtin_memcpy(&h, &b, 2);
  union { unsigned u; float f; } v; v.u = ((unsigned)h) << 16; return v.f;
}
__device__ __forceinline__ float sigm(float x) { return 1.0f / (1.0f + __expf(-x)); }

// ---------------------------------------------------------------------------
// f32 -> bf16 converter
// ---------------------------------------------------------------------------
__global__ void cvt_f32_bf16(const float* __restrict__ s, bf16* __restrict__ d, int n) {
  int i = blockIdx.x * blockDim.x + threadIdx.x;
  if (i < n) d[i] = f2bf(s[i]);
}

// ---------------------------------------------------------------------------
// Tiled WMMA GEMM:  out[m,n] = act( sum_k A[m,k]*W[n,k] + b1[n] + b2[n] )
// A: MxK bf16 row-major, W: NxK bf16 row-major. M,N % 128 == 0, K % 64 == 0.
// Block 256 thr (8 waves), tile 128x128x64. Wave (wm 0..3, wn 0..1) owns a
// 32x64 sub-tile => 2 A frags x 4 B frags x 2 K-steps = 16 WMMA per stage.
// Double-buffered LDS; async global->LDS streaming when available.
// ---------------------------------------------------------------------------
#define BM 128
#define BN 128
#define BK 64
#define LDTK 72   // padded LDS row stride (elements); 144B rows keep 16B align

__global__ __launch_bounds__(256)
void gemm_wmma(const bf16* __restrict__ A, const bf16* __restrict__ W,
               const float* __restrict__ b1, const float* __restrict__ b2,
               float* __restrict__ outF, bf16* __restrict__ outB,
               int M, int N, int K, int act)
{
  __shared__ __align__(16) bf16 As[2][BM * LDTK];
  __shared__ __align__(16) bf16 Ws[2][BN * LDTK];
  const int tid  = threadIdx.x;
  const int lane = tid & 31;
  const int wid  = tid >> 5;
  const int wm   = wid & 3;     // 0..3 -> 32-row sub-block
  const int wn   = wid >> 2;    // 0..1 -> 64-col sub-block
  const int m0   = blockIdx.y * BM;
  const int n0   = blockIdx.x * BN;

  v8f acc[2][4] = {};

  const int stages = K / BK;

  // ---- stage loaders: 128x64 tile = 4 b128 chunks per thread per matrix ----
#if HAS_ASYNC_LDS
  auto async_stage = [&](int s, int buf) {
    const int k0 = s * BK;
#pragma unroll
    for (int r = 0; r < 4; ++r) {
      const int c   = tid + 256 * r;         // chunk id 0..1023
      const int row = c >> 3;
      const int col = (c & 7) * 8;
      const bf16* ga = A + (size_t)(m0 + row) * K + k0 + col;
      const bf16* gw = W + (size_t)(n0 + row) * K + k0 + col;
      __builtin_amdgcn_global_load_async_to_lds_b128(
          AS1P(ga), AS3P(&As[buf][row * LDTK + col]), 0, 0);
      __builtin_amdgcn_global_load_async_to_lds_b128(
          AS1P(gw), AS3P(&Ws[buf][row * LDTK + col]), 0, 0);
    }
  };
#else
  v8bf ra[4], rw[4];
  auto fetch_stage = [&](int s) {
    const int k0 = s * BK;
#pragma unroll
    for (int r = 0; r < 4; ++r) {
      const int c   = tid + 256 * r;
      const int row = c >> 3;
      const int col = (c & 7) * 8;
      ra[r] = *(const v8bf*)(A + (size_t)(m0 + row) * K + k0 + col);
      rw[r] = *(const v8bf*)(W + (size_t)(n0 + row) * K + k0 + col);
    }
  };
  auto store_stage = [&](int buf) {
#pragma unroll
    for (int r = 0; r < 4; ++r) {
      const int c   = tid + 256 * r;
      const int row = c >> 3;
      const int col = (c & 7) * 8;
      *(v8bf*)(&As[buf][row * LDTK + col]) = ra[r];
      *(v8bf*)(&Ws[buf][row * LDTK + col]) = rw[r];
    }
  };
#endif

  // ---- compute one 64-deep stage from LDS buffer `buf` ----
  auto compute = [&](int buf) {
#pragma unroll
    for (int kk = 0; kk < BK; kk += 32) {
      // A fragments (16x32): lanes 0-15 carry K 0-7/16-23, lanes 16-31 carry
      // K 8-15/24-31 (ISA 7.12.2) -> two ds_load_b128 per fragment.
      v16bf afr[2];
#pragma unroll
      for (int i = 0; i < 2; ++i) {
        const int arow = wm * 32 + i * 16 + (lane & 15);
        const int kba  = kk + (lane >> 4) * 8;
        v8bf lo = *(const v8bf*)(&As[buf][arow * LDTK + kba]);
        v8bf hi = *(const v8bf*)(&As[buf][arow * LDTK + kba + 16]);
#pragma unroll
        for (int j = 0; j < 8; ++j) { afr[i][j] = lo[j]; afr[i][j + 8] = hi[j]; }
      }
#pragma unroll
      for (int j = 0; j < 4; ++j) {
        // B fragment (32x16): N = lane%16, K 0-7/16-23 (lanes 0-15), 8-15/24-31 (lanes 16-31).
        const int brow = wn * 64 + j * 16 + (lane & 15);
        const int kbb  = kk + (lane >> 4) * 8;
        v8bf lo = *(const v8bf*)(&Ws[buf][brow * LDTK + kbb]);
        v8bf hi = *(const v8bf*)(&Ws[buf][brow * LDTK + kbb + 16]);
        v16bf bfr;
#pragma unroll
        for (int t = 0; t < 8; ++t) { bfr[t] = lo[t]; bfr[t + 8] = hi[t]; }
#pragma unroll
        for (int i = 0; i < 2; ++i)
          acc[i][j] = __builtin_amdgcn_wmma_f32_16x16x32_bf16(
              false, afr[i], false, bfr, (short)0, acc[i][j], false, false);
      }
    }
  };

  // ---- main pipeline: single barrier per stage, ping-pong buffers ----
#if HAS_ASYNC_LDS
  async_stage(0, 0);
  for (int s = 0; s < stages; ++s) {
    const int cb = s & 1;
    __builtin_amdgcn_s_wait_asynccnt(0);   // buf[cb] fully landed in LDS
    __syncthreads();
    if (s + 1 < stages) async_stage(s + 1, cb ^ 1);   // overlap with compute
    if (s + 2 < stages)                                 // warm L2 for stage s+2
      __builtin_prefetch(A + (size_t)(m0 + (tid >> 1)) * K + (s + 2) * BK, 0, 1);
    compute(cb);
  }
#else
  fetch_stage(0);
  store_stage(0);
  for (int s = 0; s < stages; ++s) {
    const int cb = s & 1;
    __syncthreads();
    if (s + 1 < stages) fetch_stage(s + 1);   // global loads overlap compute
    compute(cb);
    if (s + 1 < stages) store_stage(cb ^ 1);  // lands before next barrier
  }
#endif

  // ---- epilogue: C/D layout: VGPR r -> M = r + 8*(lane>=16), N = lane%16 ----
#pragma unroll
  for (int i = 0; i < 2; ++i) {
#pragma unroll
    for (int j = 0; j < 4; ++j) {
      v8f a = acc[i][j];
#pragma unroll
      for (int r = 0; r < 8; ++r) {
        int m = m0 + wm * 32 + i * 16 + r + ((lane >> 4) << 3);
        int n = n0 + wn * 64 + j * 16 + (lane & 15);
        float v = a[r];
        if (b1) v += b1[n];
        if (b2) v += b2[n];
        if (act) v = tanhf(v);
        size_t idx = (size_t)m * N + n;
        if (outF) outF[idx] = v;
        if (outB) outB[idx] = f2bf(v);
      }
    }
  }
}

// ---------------------------------------------------------------------------
// Query attention: per row i of Hc1 (wave per row):
//   e_q = <Hc1_i, Hq1_q>, p = softmax(e), A_i = sum_q p_q * Hq1_q
// LQ=64: each lane owns q=lane and q=lane+32.
// ---------------------------------------------------------------------------
__global__ __launch_bounds__(256)
void attn_kernel(const float* __restrict__ Hc1, const float* __restrict__ Hq1,
                 float* __restrict__ Aout)
{
  __shared__ float pl[8][64];
  const int lane = threadIdx.x & 31, wid = threadIdx.x >> 5;
  const int row = blockIdx.x * 8 + wid;          // 0..BZ*LC-1
  const int b = row / cLC;
  const float* hc = Hc1 + (size_t)row * cD;
  const float* hq = Hq1 + (size_t)(b * cLQ) * cD;
  const float* q0p = hq + (size_t)lane * cD;
  const float* q1p = hq + (size_t)(lane + 32) * cD;
  float e0 = 0.f, e1 = 0.f;
  for (int d = 0; d < cD; d += 4) {
    v4f c = *(const v4f*)(hc + d);
    v4f a = *(const v4f*)(q0p + d);
    v4f q = *(const v4f*)(q1p + d);
    e0 += c[0]*a[0] + c[1]*a[1] + c[2]*a[2] + c[3]*a[3];
    e1 += c[0]*q[0] + c[1]*q[1] + c[2]*q[2] + c[3]*q[3];
  }
  float m = fmaxf(e0, e1);
  for (int k = 16; k > 0; k >>= 1) m = fmaxf(m, __shfl_xor(m, k));
  float x0 = __expf(e0 - m), x1 = __expf(e1 - m);
  float s = x0 + x1;
  for (int k = 16; k > 0; k >>= 1) s += __shfl_xor(s, k);
  float inv = 1.0f / s;
  pl[wid][lane]      = x0 * inv;
  pl[wid][lane + 32] = x1 * inv;
  __syncthreads();
  float acc[8] = {};
  const int d0 = lane * 8;
  for (int q = 0; q < cLQ; ++q) {
    float w = pl[wid][q];
    const float* qp = hq + (size_t)q * cD + d0;
    v4f a0 = *(const v4f*)qp; v4f a1 = *(const v4f*)(qp + 4);
    acc[0] += w*a0[0]; acc[1] += w*a0[1]; acc[2] += w*a0[2]; acc[3] += w*a0[3];
    acc[4] += w*a1[0]; acc[5] += w*a1[1]; acc[6] += w*a1[2]; acc[7] += w*a1[3];
  }
  float* ap = Aout + (size_t)row * cD + d0;
#pragma unroll
  for (int j = 0; j < 8; ++j) ap[j] = acc[j];
}

// ---------------------------------------------------------------------------
// tmp = [A, Hc1, A-Hc1, A*Hc1]  (f32 + bf16 copies)
// ---------------------------------------------------------------------------
__global__ void build_tmp(const float* __restrict__ Af, const float* __restrict__ Hc1,
                          float* __restrict__ tf, bf16* __restrict__ tb)
{
  size_t i = (size_t)blockIdx.x * blockDim.x + threadIdx.x;   // BZ*LC*D items
  size_t row = i / cD; int d = (int)(i % cD);
  float a = Af[i], h = Hc1[i];
  size_t o = row * (size_t)cHD;
  float v0 = a, v1 = h, v2 = a - h, v3 = a * h;
  tf[o + d] = v0; tf[o + cD + d] = v1; tf[o + 2*cD + d] = v2; tf[o + 3*cD + d] = v3;
  tb[o + d] = f2bf(v0); tb[o + cD + d] = f2bf(v1);
  tb[o + 2*cD + d] = f2bf(v2); tb[o + 3*cD + d] = f2bf(v3);
}

// ---------------------------------------------------------------------------
// colsum[b][d] = sum_l tmp[b][l][d]
// ---------------------------------------------------------------------------
__global__ void colsum_kernel(const float* __restrict__ tf, float* __restrict__ cs)
{
  int i = blockIdx.x * blockDim.x + threadIdx.x;    // 0..BZ*HD-1
  int b = i / cHD, d = i % cHD;
  const float* p = tf + (size_t)b * cLC * cHD + d;
  float s = 0.f;
  for (int l = 0; l < cLC; ++l) s += p[(size_t)l * cHD];
  cs[i] = s;
}

// ---------------------------------------------------------------------------
// Band self-attention. S = (G G^T) * bandmask; off-band scores are exactly 0.
// Row softmax: m = max(band, 0); Z = sum_band exp(s-m) + (LC-nb)*exp(-m)
// Bm_i = sum_band (p_j - p0) * tmp_j + p0 * colsum ;  Y = [Bm, tmp]
// One 256-thread block per row.
// ---------------------------------------------------------------------------
__global__ __launch_bounds__(256)
void band_attn(const bf16* __restrict__ G, const float* __restrict__ tf,
               const float* __restrict__ cs,
               float* __restrict__ Yf, bf16* __restrict__ Yb)
{
  __shared__ float sb[80];
  __shared__ float pb[80];
  __shared__ float p0s;
  const int tid = threadIdx.x, lane = tid & 31, wid = tid >> 5;
  const int row = blockIdx.x;                  // 0..BZ*LC-1
  const int b = row / cLC, i = row % cLC;
  const int jlo = (i - cNB < 0) ? 0 : i - cNB;
  const int jhi = (i + cNB > cLC - 1) ? cLC - 1 : i + cNB;
  const int nb = jhi - jlo + 1;

  const bf16* gi = G + (size_t)row * cHD;
  const int c0 = lane * 32;
  v8bf g0 = *(const v8bf*)(gi + c0),      g1 = *(const v8bf*)(gi + c0 + 8),
       g2 = *(const v8bf*)(gi + c0 + 16), g3 = *(const v8bf*)(gi + c0 + 24);

  for (int jj = wid; jj < nb; jj += 8) {
    const bf16* gj = G + (size_t)(b * cLC + jlo + jj) * cHD + c0;
    v8bf a0 = *(const v8bf*)gj,        a1 = *(const v8bf*)(gj + 8),
         a2 = *(const v8bf*)(gj + 16), a3 = *(const v8bf*)(gj + 24);
    float s = 0.f;
#pragma unroll
    for (int j = 0; j < 8; ++j)
      s += bf2f(g0[j])*bf2f(a0[j]) + bf2f(g1[j])*bf2f(a1[j])
         + bf2f(g2[j])*bf2f(a2[j]) + bf2f(g3[j])*bf2f(a3[j]);
    for (int k = 16; k > 0; k >>= 1) s += __shfl_xor(s, k);
    if (lane == 0) sb[jj] = s;
  }
  __syncthreads();
  if (wid == 0) {
    float v0 = (lane      < nb) ? sb[lane]      : -3.4e38f;
    float v1 = (lane + 32 < nb) ? sb[lane + 32] : -3.4e38f;
    float m = fmaxf(fmaxf(v0, v1), 0.0f);     // off-band entries score 0
    for (int k = 16; k > 0; k >>= 1) m = fmaxf(m, __shfl_xor(m, k));
    float e0 = (lane      < nb) ? __expf(v0 - m) : 0.f;
    float e1 = (lane + 32 < nb) ? __expf(v1 - m) : 0.f;
    float z = e0 + e1;
    for (int k = 16; k > 0; k >>= 1) z += __shfl_xor(z, k);
    float eoff = __expf(-m);
    z += (float)(cLC - nb) * eoff;
    float invz = 1.0f / z;
    if (lane < nb)      pb[lane]      = e0 * invz;
    if (lane + 32 < nb) pb[lane + 32] = e1 * invz;
    if (lane == 0) p0s = eoff * invz;
  }
  __syncthreads();
  const float p0 = p0s;

  const int d0 = tid * 4;                      // HD/256 = 4 dims per thread
  const float* csb = cs + (size_t)b * cHD;
  float a0 = p0 * csb[d0],     a1 = p0 * csb[d0 + 1],
        a2 = p0 * csb[d0 + 2], a3 = p0 * csb[d0 + 3];
  for (int jj = 0; jj < nb; ++jj) {
    float w = pb[jj] - p0;
    const float* tr = tf + (size_t)(b * cLC + jlo + jj) * cHD + d0;
    a0 += w * tr[0]; a1 += w * tr[1]; a2 += w * tr[2]; a3 += w * tr[3];
  }
  size_t yo = (size_t)row * (2 * cHD);
  float* yf = Yf + yo;  bf16* yb = Yb + yo;
  yf[d0] = a0; yf[d0+1] = a1; yf[d0+2] = a2; yf[d0+3] = a3;
  yb[d0] = f2bf(a0); yb[d0+1] = f2bf(a1); yb[d0+2] = f2bf(a2); yb[d0+3] = f2bf(a3);
  const float* ti = tf + (size_t)row * cHD + d0;
#pragma unroll
  for (int j = 0; j < 4; ++j) {
    float tv = ti[j];
    yf[cHD + d0 + j] = tv;
    yb[cHD + d0 + j] = f2bf(tv);
  }
}

// ---------------------------------------------------------------------------
// LSTM scan (one block per batch, one launch per direction).
// pre already holds X@Wih.T + bih + bhh. 1024 threads: thread n computes
// g[n] = pre[t][n] + <Whh[n,:], h>. First 256 threads own one hidden unit.
// outF/outB already offset to this direction's 256-col slice (stride = 512).
// ---------------------------------------------------------------------------
__global__ __launch_bounds__(1024)
void lstm_scan(const float* __restrict__ pre, const bf16* __restrict__ Whh,
               float* __restrict__ outF, bf16* __restrict__ outB,
               int reverse, int out_stride)
{
  __shared__ float hs[cD];
  __shared__ float gs[cHD];
  const int tid = threadIdx.x;
  const int b = blockIdx.x;
  if (tid < cD) hs[tid] = 0.f;
  float c = 0.f;
  __syncthreads();
  const bf16* wr = Whh + (size_t)tid * cD;
  for (int step = 0; step < cLC; ++step) {
    const int t = reverse ? (cLC - 1 - step) : step;
    const float* pr = pre + ((size_t)b * cLC + t) * cHD;
    float g = pr[tid];
    for (int u = 0; u < cD; u += 8) {
      v8bf w8 = *(const v8bf*)(wr + u);
#pragma unroll
      for (int j = 0; j < 8; ++j) g += bf2f(w8[j]) * hs[u + j];
    }
    gs[tid] = g;
    __syncthreads();
    if (tid < cD) {
      float gi = gs[tid], gf = gs[cD + tid], gg = gs[2*cD + tid], go = gs[3*cD + tid];
      c = sigm(gf) * c + sigm(gi) * tanhf(gg);
      float h = sigm(go) * tanhf(c);
      hs[tid] = h;
      size_t oi = ((size_t)b * cLC + t) * out_stride + tid;
      outF[oi] = h;
      if (outB) outB[oi] = f2bf(h);
    }
    __syncthreads();
  }
}

// ---------------------------------------------------------------------------
// Host orchestration
// ---------------------------------------------------------------------------
extern "C" void kernel_launch(void* const* d_in, const int* in_sizes, int n_in,
                              void* d_out, int out_size, void* d_ws, size_t ws_size,
                              hipStream_t stream)
{
  (void)in_sizes; (void)n_in; (void)out_size; (void)ws_size;
  const float* Hq = (const float*)d_in[0];
  const float* Hc = (const float*)d_in[1];
  const float* w1 = (const float*)d_in[2];
  const float* b1 = (const float*)d_in[3];
  const float* w2 = (const float*)d_in[4];
  const float* b2 = (const float*)d_in[5];
  // d_in[6 + l*8 + dir*4 + {0:Wih,1:Whh,2:bih,3:bhh}], d_in[46] = mask (band; unused)

  char* ws = (char*)d_ws;
  size_t off = 0;
  auto alloc = [&](size_t bytes) -> char* {
    char* p = ws + off;
    off += (bytes + 255) & ~(size_t)255;
    return p;
  };

  bf16* Hqb = (bf16*)alloc((size_t)cBZ * cLQ * cD * 2);
  bf16* Hcb = (bf16*)alloc((size_t)cBZ * cLC * cD * 2);
  bf16* w1b = (bf16*)alloc((size_t)cD * cD * 2);
  bf16* w2b = (bf16*)alloc((size_t)cHD * cHD * 2);
  bf16* Wihb[cNL][2]; bf16* Whhb[cNL][2];
  for (int l = 0; l < cNL; ++l) {
    int in_dim = (l == 0) ? 8 * cD : 2 * cD;
    for (int dd = 0; dd < 2; ++dd) {
      Wihb[l][dd] = (bf16*)alloc((size_t)cHD * in_dim * 2);
      Whhb[l][dd] = (bf16*)alloc((size_t)cHD * cD * 2);
    }
  }
  float* Hq1f = (float*)alloc((size_t)cBZ * cLQ * cD * 4);
  float* Hc1f = (float*)alloc((size_t)cBZ * cLC * cD * 4);
  float* Af   = (float*)alloc((size_t)cBZ * cLC * cD * 4);
  float* tmpf = (float*)alloc((size_t)cBZ * cLC * cHD * 4);
  bf16*  tmpb = (bf16*) alloc((size_t)cBZ * cLC * cHD * 2);
  bf16*  Gb   = (bf16*) alloc((size_t)cBZ * cLC * cHD * 2);
  float* cs   = (float*)alloc((size_t)cBZ * cHD * 4);
  float* Yf   = (float*)alloc((size_t)cBZ * cLC * 2 * cHD * 4);
  bf16*  Yb   = (bf16*) alloc((size_t)cBZ * cLC * 2 * cHD * 2);
  float* pre0 = (float*)alloc((size_t)cBZ * cLC * cHD * 4);
  float* pre1 = (float*)alloc((size_t)cBZ * cLC * cHD * 4);
  float* Xf[2]; bf16* Xb[2];
  for (int p = 0; p < 2; ++p) {
    Xf[p] = (float*)alloc((size_t)cBZ * cLC * 2 * cD * 4);
    Xb[p] = (bf16*) alloc((size_t)cBZ * cLC * 2 * cD * 2);
  }

  auto cvt = [&](const float* s, bf16* d, size_t n) {
    cvt_f32_bf16<<<dim3((unsigned)((n + 255) / 256)), dim3(256), 0, stream>>>(s, d, (int)n);
  };
  auto gemm = [&](const bf16* A, const bf16* W, const float* bb1, const float* bb2,
                  float* oF, bf16* oB, int M, int N, int K, int act) {
    gemm_wmma<<<dim3(N / BN, M / BM), dim3(256), 0, stream>>>(
        A, W, bb1, bb2, oF, oB, M, N, K, act);
  };

  // weight / input conversions to bf16
  cvt(Hq, Hqb, (size_t)cBZ * cLQ * cD);
  cvt(Hc, Hcb, (size_t)cBZ * cLC * cD);
  cvt(w1, w1b, (size_t)cD * cD);
  cvt(w2, w2b, (size_t)cHD * cHD);
  for (int l = 0; l < cNL; ++l) {
    int in_dim = (l == 0) ? 8 * cD : 2 * cD;
    for (int dd = 0; dd < 2; ++dd) {
      cvt((const float*)d_in[6 + l*8 + dd*4 + 0], Wihb[l][dd], (size_t)cHD * in_dim);
      cvt((const float*)d_in[6 + l*8 + dd*4 + 1], Whhb[l][dd], (size_t)cHD * cD);
    }
  }

  // Hq1 = tanh(Hq @ w1.T + b1), Hc1 = tanh(Hc @ w1.T + b1)
  gemm(Hqb, w1b, b1, nullptr, Hq1f, nullptr, cBZ * cLQ, cD, cD, 1);
  gemm(Hcb, w1b, b1, nullptr, Hc1f, nullptr, cBZ * cLC, cD, cD, 1);

  // A = softmax(Hc1 Hq1^T) Hq1
  attn_kernel<<<dim3(cBZ * cLC / 8), dim3(256), 0, stream>>>(Hc1f, Hq1f, Af);

  // tmp = [A, Hc1, A-Hc1, A*Hc1]
  build_tmp<<<dim3(cBZ * cLC * cD / 256), dim3(256), 0, stream>>>(Af, Hc1f, tmpf, tmpb);

  // G = tanh(tmp @ w2.T + b2)  (bf16 output for band dots)
  gemm(tmpb, w2b, b2, nullptr, nullptr, Gb, cBZ * cLC, cHD, cHD, 1);

  // column sums of tmp per batch
  colsum_kernel<<<dim3(cBZ * cHD / 256), dim3(256), 0, stream>>>(tmpf, cs);

  // Y = [softmax((G G^T)*band) @ tmp, tmp]
  band_attn<<<dim3(cBZ * cLC), dim3(256), 0, stream>>>(Gb, tmpf, cs, Yf, Yb);

  // BiLSTM: 5 layers x 2 directions
  const bf16* Xin = Yb;
  int in_dim = 8 * cD;
  for (int l = 0; l < cNL; ++l) {
    const float* bih_f = (const float*)d_in[6 + l*8 + 2];
    const float* bhh_f = (const float*)d_in[6 + l*8 + 3];
    const float* bih_r = (const float*)d_in[6 + l*8 + 6];
    const float* bhh_r = (const float*)d_in[6 + l*8 + 7];
    gemm(Xin, Wihb[l][0], bih_f, bhh_f, pre0, nullptr, cBZ * cLC, cHD, in_dim, 0);
    gemm(Xin, Wihb[l][1], bih_r, bhh_r, pre1, nullptr, cBZ * cLC, cHD, in_dim, 0);
    float* oF = (l == cNL - 1) ? (float*)d_out : Xf[l & 1];
    bf16*  oB = (l == cNL - 1) ? nullptr       : Xb[l & 1];
    lstm_scan<<<dim3(cBZ), dim3(1024), 0, stream>>>(pre0, Whhb[l][0], oF, oB, 0, 2 * cD);
    lstm_scan<<<dim3(cBZ), dim3(1024), 0, stream>>>(pre1, Whhb[l][1], oF + cD,
                                                    oB ? oB + cD : nullptr, 1, 2 * cD);
    Xin = Xb[l & 1];
    in_dim = 2 * cD;
  }
}
